// Layer_88854283420047
// MI455X (gfx1250) — matrix operand order repaired and checked by reading, not compiled, not run on hardware
//
#include <hip/hip_runtime.h>

// Problem sizes (fixed by reference)
#define S_ 8192
#define D_ 2048
#define H_ 2048
#define F_ 8192
#define C_GATE 8.0f
#define EPS_ 1e-6f

// Scan chunking: 64 chunks x 128 steps = 8192
#define NCH 64
#define CL  128

// GEMM tiling: block 128x128, 8 waves of 32x64 (2x4 WMMA accums), K-step 64
#define BM 128
#define BN 128
#define BK 64
#define LDSK (BK + 8)   // padded LDS row stride (bf16 elems)

typedef __bf16 bf16_t;
typedef bf16_t bf16x16 __attribute__((ext_vector_type(16)));
typedef float  f32x8   __attribute__((ext_vector_type(8)));
typedef int    v4i_t   __attribute__((ext_vector_type(4)));

union Frag { bf16x16 v; uint4 q[2]; };

// ---- optional gfx1250 async global->LDS path --------------------------------
#if defined(__has_builtin)
#  if __has_builtin(__builtin_amdgcn_global_load_async_to_lds_b128)
#    define ASYNC_LDS 1
#  endif
#  if __has_builtin(__builtin_amdgcn_s_wait_asynccnt)
#    define WAIT_ASYNC() __builtin_amdgcn_s_wait_asynccnt(0)
#  else
#    define WAIT_ASYNC() asm volatile("s_wait_asynccnt 0x0" ::: "memory")
#  endif
#endif
#ifndef WAIT_ASYNC
#  define WAIT_ASYNC() ((void)0)
#endif

__device__ __forceinline__ unsigned short f2bf(float f) {
  unsigned int u = __float_as_uint(f);
  unsigned int r = u + 0x7FFFu + ((u >> 16) & 1u);   // round-to-nearest-even
  return (unsigned short)(r >> 16);
}
__device__ __forceinline__ float sigm(float x) { return 1.f / (1.f + __expf(-x)); }
__device__ __forceinline__ float log_sigmoid(float x) {
  return fminf(x, 0.f) - log1pf(__expf(-fabsf(x)));
}
__device__ __forceinline__ float gelu_tanh(float x) {
  float x3 = x * x * x;
  return 0.5f * x * (1.f + tanhf(0.7978845608028654f * (x + 0.044715f * x3)));
}

// ---------------------------------------------------------------------------
// fp32 -> bf16 conversion (grid-stride)
// ---------------------------------------------------------------------------
__global__ void cvt_bf16_kernel(const float* __restrict__ in,
                                unsigned short* __restrict__ out, size_t n) {
  size_t i = (size_t)blockIdx.x * blockDim.x + threadIdx.x;
  size_t stride = (size_t)gridDim.x * blockDim.x;
  for (; i < n; i += stride) out[i] = f2bf(in[i]);
}

// ---------------------------------------------------------------------------
// bf16 WMMA GEMM: C[M,N] = A[M,K] @ B[N,K]^T (+bias), fused epilogues
//   mode 0: f32 out           mode 1: sigmoid -> f32
//   mode 2: log_a -> f32      mode 3: gelu -> bf16 out
// 256 threads = 8 waves; block tile 128x128; wave tile 32x64 (2x4 WMMA).
// Per staged K=64 tile per wave: 12 fragment loads feed 16 WMMAs.
// Staging uses gfx1250 GLOBAL_LOAD_ASYNC_TO_LDS_B128 when available.
// ---------------------------------------------------------------------------
__global__ __launch_bounds__(256)
void gemm_bf16_kernel(const unsigned short* __restrict__ A,
                      const unsigned short* __restrict__ B,
                      const float* __restrict__ bias,
                      const float* __restrict__ lam,
                      float* __restrict__ Cf,
                      unsigned short* __restrict__ Cbf,
                      int M, int N, int K, int mode) {
  __shared__ unsigned short As[BM][LDSK];
  __shared__ unsigned short Bs[BN][LDSK];

  const int tid  = threadIdx.x;
  const int lane = tid & 31;           // wave32
  const int wave = tid >> 5;           // 0..7
  const int wm = wave & 3;             // row group (x32)
  const int wn = wave >> 2;            // col group (x64)
  const int lr = lane & 15, hi = lane >> 4;
  const int bM = blockIdx.y * BM;
  const int bN = blockIdx.x * BN;

  f32x8 acc[2][4];
#pragma unroll
  for (int i = 0; i < 2; ++i)
#pragma unroll
    for (int j = 0; j < 4; ++j) acc[i][j] = (f32x8){0, 0, 0, 0, 0, 0, 0, 0};

  const int ldrow = tid >> 3;          // 0..31 (8 x 16B chunks per 64-elem row)
  const int ldcol = (tid & 7) * 8;     // 0..56 step 8 (bf16 elems)
  const int ktiles = K / BK;

  for (int kt = 0; kt < ktiles; ++kt) {
    const int k0 = kt * BK;
    // Stage A (128x64 bf16 = 1024 x 16B) and B (same): 4+4 chunks per thread
#pragma unroll
    for (int rep = 0; rep < 4; ++rep) {
      const int row = ldrow + rep * 32;
      const size_t ga = (size_t)(bM + row) * K + k0 + ldcol;
      const size_t gb = (size_t)(bN + row) * K + k0 + ldcol;
#ifdef ASYNC_LDS
      __builtin_amdgcn_global_load_async_to_lds_b128(
          (__attribute__((address_space(1))) v4i_t*)(A + ga),
          (__attribute__((address_space(3))) v4i_t*)&As[row][ldcol], 0, 0);
      __builtin_amdgcn_global_load_async_to_lds_b128(
          (__attribute__((address_space(1))) v4i_t*)(B + gb),
          (__attribute__((address_space(3))) v4i_t*)&Bs[row][ldcol], 0, 0);
#else
      *(uint4*)&As[row][ldcol] = *(const uint4*)(A + ga);
      *(uint4*)&Bs[row][ldcol] = *(const uint4*)(B + gb);
#endif
    }
#ifndef ASYNC_LDS
    if (kt + 1 < ktiles) {
      __builtin_prefetch(A + (size_t)(bM + ldrow) * K + k0 + BK + ldcol, 0, 3);
      __builtin_prefetch(B + (size_t)(bN + ldrow) * K + k0 + BK + ldcol, 0, 3);
    }
#endif
    WAIT_ASYNC();
    __syncthreads();

    // Two K=32 sub-steps per staged tile
#pragma unroll
    for (int ks = 0; ks < 2; ++ks) {
      Frag af[2], bfr[4];
#pragma unroll
      for (int mi = 0; mi < 2; ++mi) {
        const unsigned short* ap = &As[wm * 32 + mi * 16 + lr][ks * 32];
        af[mi].q[0] = *(const uint4*)(ap + hi * 8);
        af[mi].q[1] = *(const uint4*)(ap + 16 + hi * 8);
      }
#pragma unroll
      for (int ni = 0; ni < 4; ++ni) {
        const unsigned short* bp = &Bs[wn * 64 + ni * 16 + lr][ks * 32 + hi * 8];
        bfr[ni].q[0] = *(const uint4*)(bp);
        bfr[ni].q[1] = *(const uint4*)(bp + 16);
      }
#pragma unroll
      for (int mi = 0; mi < 2; ++mi)
#pragma unroll
        for (int ni = 0; ni < 4; ++ni)
          acc[mi][ni] = __builtin_amdgcn_wmma_f32_16x16x32_bf16(
              false, af[mi].v, false, bfr[ni].v, (short)0, acc[mi][ni], false,
              false);
    }
    __syncthreads();
  }

  __builtin_amdgcn_s_wait_tensorcnt(0);

  // Epilogue. C/D layout: VGPR r -> (M = r + 8*hi, N = lr) within 16x16 tile.
#pragma unroll
  for (int mi = 0; mi < 2; ++mi) {
#pragma unroll
    for (int ni = 0; ni < 4; ++ni) {
      const int n  = bN + wn * 64 + ni * 16 + lr;
      const int m0 = bM + wm * 32 + mi * 16 + hi * 8;
      const float bv = bias ? bias[n] : 0.f;
      float ls = 0.f;
      if (mode == 2) ls = log_sigmoid(lam[n]);
#pragma unroll
      for (int r = 0; r < 8; ++r) {
        float val = acc[mi][ni][r] + bv;
        size_t off = (size_t)(m0 + r) * N + n;
        if (mode == 0)      Cf[off] = val;
        else if (mode == 1) Cf[off] = sigm(val);
        else if (mode == 2) Cf[off] = C_GATE * sigm(val) * ls;
        else                Cbf[off] = f2bf(gelu_tanh(val));
      }
    }
  }
}

// ---------------------------------------------------------------------------
// a_t = exp(log_a); b_t = sqrt(1 - exp(2 log_a)) * g * xh   (in-place)
// ---------------------------------------------------------------------------
__global__ void ab_kernel(float* __restrict__ loga_to_a,
                          float* __restrict__ g_to_b,
                          const float* __restrict__ xh, size_t n) {
  size_t i = (size_t)blockIdx.x * blockDim.x + threadIdx.x;
  size_t stride = (size_t)gridDim.x * blockDim.x;
  for (; i < n; i += stride) {
    float la = loga_to_a[i];
    float a  = __expf(la);
    float em = __expf(2.f * la);
    float b  = sqrtf(fmaxf(1.f - em, 0.f)) * g_to_b[i] * xh[i];
    loga_to_a[i] = a;
    g_to_b[i]    = b;
  }
}

// ---------------------------------------------------------------------------
// Chunked linear-recurrence scan: h_t = a_t * h_{t-1} + b_t per channel
// ---------------------------------------------------------------------------
__global__ void scan1_kernel(const float* __restrict__ a,
                             const float* __restrict__ b,
                             float* __restrict__ Aagg,
                             float* __restrict__ Bagg) {
  int idx = blockIdx.x * blockDim.x + threadIdx.x;  // NCH*H threads
  int c = idx & (H_ - 1);
  int ch = idx >> 11;                               // /H_
  float pa = 1.f, hb = 0.f;
  size_t base = (size_t)ch * CL * H_ + c;
  for (int i = 0; i < CL; ++i) {
    float ai = a[base], bi = b[base];
    pa *= ai;
    hb = ai * hb + bi;
    base += H_;
  }
  Aagg[ch * H_ + c] = pa;
  Bagg[ch * H_ + c] = hb;
}

__global__ void scan2_kernel(float* __restrict__ Aagg,
                             const float* __restrict__ Bagg) {
  int c = blockIdx.x * blockDim.x + threadIdx.x;    // H_ threads
  if (c >= H_) return;
  float carry = 0.f;
  for (int ch = 0; ch < NCH; ++ch) {
    float pa = Aagg[ch * H_ + c];
    float hb = Bagg[ch * H_ + c];
    Aagg[ch * H_ + c] = carry;                      // prefix carry before chunk
    carry = pa * carry + hb;
  }
}

__global__ void scan3_kernel(const float* __restrict__ a,
                             const float* __restrict__ b,
                             const float* __restrict__ pref,
                             unsigned short* __restrict__ hbf) {
  int idx = blockIdx.x * blockDim.x + threadIdx.x;
  int c = idx & (H_ - 1);
  int ch = idx >> 11;
  float h = pref[ch * H_ + c];
  size_t base = (size_t)ch * CL * H_ + c;
  for (int i = 0; i < CL; ++i) {
    h = a[base] * h + b[base];
    hbf[base] = f2bf(h);
    base += H_;
  }
}

// ---------------------------------------------------------------------------
// Stochastic gate + residual + RMSNorm, one block per row (256 threads, D=2048)
// ---------------------------------------------------------------------------
__global__ __launch_bounds__(256)
void gate_norm_kernel(const float* __restrict__ xin,
                      const float* __restrict__ resid,
                      const float* __restrict__ actw,
                      const float* __restrict__ actb,
                      const float* __restrict__ u,
                      const float* __restrict__ nw,
                      const float* __restrict__ nb,
                      float* __restrict__ xout,
                      unsigned short* __restrict__ xout_bf) {
  __shared__ float red[256];
  const int s = blockIdx.x, tid = threadIdx.x;
  const float* xr = xin + (size_t)s * D_;
  const float* rr = resid + (size_t)s * D_;

  float dot = 0.f;
#pragma unroll
  for (int i = 0; i < 8; ++i) {
    int d = tid + i * 256;
    dot += rr[d] * actw[d];
  }
  red[tid] = dot;
  __syncthreads();
  for (int off = 128; off > 0; off >>= 1) {
    if (tid < off) red[tid] += red[tid + off];
    __syncthreads();
  }
  const float prob = sigm(red[0] + actb[0]);
  const float keep = (u[s] < prob) ? 1.f : 0.f;
  __syncthreads();

  float v[8];
  float ss = 0.f;
#pragma unroll
  for (int i = 0; i < 8; ++i) {
    int d = tid + i * 256;
    float t = xr[d] + keep * rr[d];
    v[i] = t;
    ss += t * t;
  }
  red[tid] = ss;
  __syncthreads();
  for (int off = 128; off > 0; off >>= 1) {
    if (tid < off) red[tid] += red[tid + off];
    __syncthreads();
  }
  const float inv = rsqrtf(red[0] * (1.f / (float)D_) + EPS_);
#pragma unroll
  for (int i = 0; i < 8; ++i) {
    int d = tid + i * 256;
    float o = v[i] * inv * nw[d] + nb[d];
    xout[(size_t)s * D_ + d] = o;
    if (xout_bf) xout_bf[(size_t)s * D_ + d] = f2bf(o);
  }
}

// ---------------------------------------------------------------------------
extern "C" void kernel_launch(void* const* d_in, const int* in_sizes, int n_in,
                              void* d_out, int out_size, void* d_ws,
                              size_t ws_size, hipStream_t stream) {
  (void)in_sizes; (void)n_in; (void)out_size; (void)ws_size;

  const float* x     = (const float*)d_in[0];
  const float* W_in  = (const float*)d_in[1];
  const float* b_in  = (const float*)d_in[2];
  const float* W_a   = (const float*)d_in[3];
  const float* b_a   = (const float*)d_in[4];
  const float* W_g   = (const float*)d_in[5];
  const float* b_g   = (const float*)d_in[6];
  const float* lam   = (const float*)d_in[7];
  const float* W_out = (const float*)d_in[8];
  const float* b_out = (const float*)d_in[9];
  const float* w1    = (const float*)d_in[10];
  const float* fb1   = (const float*)d_in[11];
  const float* w2    = (const float*)d_in[12];
  const float* fb2   = (const float*)d_in[13];
  const float* a1w   = (const float*)d_in[14];
  const float* a1b   = (const float*)d_in[15];
  const float* a2w   = (const float*)d_in[16];
  const float* a2b   = (const float*)d_in[17];
  const float* n1w   = (const float*)d_in[18];
  const float* n1b   = (const float*)d_in[19];
  const float* n2w   = (const float*)d_in[20];
  const float* n2b   = (const float*)d_in[21];
  const float* u1    = (const float*)d_in[22];
  const float* u2    = (const float*)d_in[23];

  // Workspace carve (256B aligned)
  char* cur = (char*)d_ws;
  auto carve = [&](size_t bytes) -> void* {
    void* p = (void*)cur;
    cur += (bytes + 255) & ~(size_t)255;
    return p;
  };
  unsigned short* xbf   = (unsigned short*)carve((size_t)S_ * D_ * 2);
  unsigned short* winb  = (unsigned short*)carve((size_t)H_ * D_ * 2);
  unsigned short* wab   = (unsigned short*)carve((size_t)H_ * D_ * 2);
  unsigned short* wgb   = (unsigned short*)carve((size_t)H_ * D_ * 2);
  unsigned short* woutb = (unsigned short*)carve((size_t)D_ * H_ * 2);
  unsigned short* w1b   = (unsigned short*)carve((size_t)F_ * D_ * 2);
  unsigned short* w2b   = (unsigned short*)carve((size_t)D_ * F_ * 2);
  float* xh    = (float*)carve((size_t)S_ * H_ * 4);
  float* abuf  = (float*)carve((size_t)S_ * H_ * 4);   // log_a -> a
  float* bbuf  = (float*)carve((size_t)S_ * H_ * 4);   // g -> b
  float* Aagg  = (float*)carve((size_t)NCH * H_ * 4);
  float* Bagg  = (float*)carve((size_t)NCH * H_ * 4);
  unsigned short* hbf = (unsigned short*)carve((size_t)S_ * H_ * 2);
  float* resid = (float*)carve((size_t)S_ * D_ * 4);
  float* x1    = (float*)carve((size_t)S_ * D_ * 4);
  unsigned short* x1bf = (unsigned short*)carve((size_t)S_ * D_ * 2);
  unsigned short* tbf  = (unsigned short*)carve((size_t)S_ * F_ * 2);

  const dim3 blk(256);

  // --- bf16 conversions -----------------------------------------------------
  cvt_bf16_kernel<<<4096, blk, 0, stream>>>(x, xbf, (size_t)S_ * D_);
  cvt_bf16_kernel<<<2048, blk, 0, stream>>>(W_in, winb, (size_t)H_ * D_);
  cvt_bf16_kernel<<<2048, blk, 0, stream>>>(W_a, wab, (size_t)H_ * D_);
  cvt_bf16_kernel<<<2048, blk, 0, stream>>>(W_g, wgb, (size_t)H_ * D_);
  cvt_bf16_kernel<<<2048, blk, 0, stream>>>(W_out, woutb, (size_t)D_ * H_);
  cvt_bf16_kernel<<<4096, blk, 0, stream>>>(w1, w1b, (size_t)F_ * D_);
  cvt_bf16_kernel<<<4096, blk, 0, stream>>>(w2, w2b, (size_t)D_ * F_);

  // --- RG-LRU input projections (3 WMMA GEMMs) ------------------------------
  dim3 gSH(H_ / BN, S_ / BM);  // (16, 64)
  gemm_bf16_kernel<<<gSH, blk, 0, stream>>>(xbf, winb, b_in, nullptr, xh,
                                            nullptr, S_, H_, D_, 0);
  gemm_bf16_kernel<<<gSH, blk, 0, stream>>>(xbf, wab, b_a, lam, abuf, nullptr,
                                            S_, H_, D_, 2);
  gemm_bf16_kernel<<<gSH, blk, 0, stream>>>(xbf, wgb, b_g, nullptr, bbuf,
                                            nullptr, S_, H_, D_, 1);

  // --- gates -> (a, b) ------------------------------------------------------
  ab_kernel<<<65536, blk, 0, stream>>>(abuf, bbuf, xh, (size_t)S_ * H_);

  // --- chunked associative scan h_t = a_t h_{t-1} + b_t ---------------------
  scan1_kernel<<<(NCH * H_) / 256, blk, 0, stream>>>(abuf, bbuf, Aagg, Bagg);
  scan2_kernel<<<H_ / 256, blk, 0, stream>>>(Aagg, Bagg);
  scan3_kernel<<<(NCH * H_) / 256, blk, 0, stream>>>(abuf, bbuf, Aagg, hbf);

  // --- output projection ----------------------------------------------------
  dim3 gSD(D_ / BN, S_ / BM);  // (16, 64)
  gemm_bf16_kernel<<<gSD, blk, 0, stream>>>(hbf, woutb, b_out, nullptr, resid,
                                            nullptr, S_, D_, H_, 0);

  // --- gate 1 + residual + RMSNorm ------------------------------------------
  gate_norm_kernel<<<S_, blk, 0, stream>>>(x, resid, a1w, a1b, u1, n1w, n1b,
                                           x1, x1bf);

  // --- FFN: gelu(x1 @ w1^T + b1) @ w2^T + b2 --------------------------------
  dim3 gSF(F_ / BN, S_ / BM);  // (64, 64)
  gemm_bf16_kernel<<<gSF, blk, 0, stream>>>(x1bf, w1b, fb1, nullptr, nullptr,
                                            tbf, S_, F_, D_, 3);
  gemm_bf16_kernel<<<gSD, blk, 0, stream>>>(tbf, w2b, fb2, nullptr, resid,
                                            nullptr, S_, D_, F_, 0);

  // --- gate 2 + residual + RMSNorm -> final output --------------------------
  gate_norm_kernel<<<S_, blk, 0, stream>>>(x1, resid, a2w, a2b, u2, n2w, n2b,
                                           (float*)d_out, nullptr);
}
